// self_attention_layer_74749610819968
// MI455X (gfx1250) — matrix operand order repaired and checked by reading, not compiled, not run on hardware
//
#include <hip/hip_runtime.h>
#include <hip/hip_bf16.h>
#include <hip/hip_fp16.h>

#define BB   2
#define SS   2048
#define DD   1024
#define HH   16
#define DHD  64
#define MTOT (BB*SS)          // 4096 rows
#define LOG2E 1.44269504088896f
#define SCL   (0.125f * LOG2E)   // 1/sqrt(DH) folded with log2(e)

typedef __attribute__((ext_vector_type(16))) _Float16 v16h;
typedef __attribute__((ext_vector_type(8)))  _Float16 v8h;
typedef __attribute__((ext_vector_type(8)))  float    v8f;

// ---------------------------------------------------------------- helpers ---
static __device__ __forceinline__ v16h make_frag(v8h lo, v8h hi) {
  v16h f;
#pragma unroll
  for (int i = 0; i < 8; ++i) { f[i] = lo[i]; f[i + 8] = hi[i]; }
  return f;
}
// A 16x32 f16, this code's k order (not the ISA table's): per-lane 16 contiguous K values (k0..k0+15)
static __device__ __forceinline__ v16h load_frag_split(const _Float16* p) {
  return make_frag(*(const v8h*)(p), *(const v8h*)(p + 8));
}
// B 32x16 f16, same k order as A here (not the ISA table's): per-lane 16 contiguous K values
static __device__ __forceinline__ v16h load_frag_contig(const _Float16* p) {
  return make_frag(*(const v8h*)(p), *(const v8h*)(p + 8));
}
static __device__ __forceinline__ v8f wmma_f16(v16h a, v16h b, v8f c) {
  return __builtin_amdgcn_wmma_f32_16x16x32_f16(false, a, false, b, (short)0, c,
                                                false, false);
}

// 16-lane xor-butterfly reductions via DPP16 ROW_XMASK (VALU, no LDS traffic).
#if __has_builtin(__builtin_amdgcn_update_dpp)
template <int M>
static __device__ __forceinline__ float row_xor_f(float v) {
  return __int_as_float(__builtin_amdgcn_update_dpp(
      0, __float_as_int(v), 0x160 + M, 0xf, 0xf, true));
}
static __device__ __forceinline__ float rowmax16(float v) {
  v = fmaxf(v, row_xor_f<1>(v));
  v = fmaxf(v, row_xor_f<2>(v));
  v = fmaxf(v, row_xor_f<4>(v));
  v = fmaxf(v, row_xor_f<8>(v));
  return v;
}
static __device__ __forceinline__ float rowsum16(float v) {
  v += row_xor_f<1>(v);
  v += row_xor_f<2>(v);
  v += row_xor_f<4>(v);
  v += row_xor_f<8>(v);
  return v;
}
#else
static __device__ __forceinline__ float rowmax16(float v) {
  for (int m = 1; m < 16; m <<= 1) v = fmaxf(v, __shfl_xor(v, m, 32));
  return v;
}
static __device__ __forceinline__ float rowsum16(float v) {
  for (int m = 1; m < 16; m <<= 1) v += __shfl_xor(v, m, 32);
  return v;
}
#endif

// ------------------------------------------------------------ conversions ---
__global__ void cvt_f32_to_f16(const float* __restrict__ x,
                               _Float16* __restrict__ y, int n) {
  int i = blockIdx.x * 256 + threadIdx.x;
  if (i < n) y[i] = (_Float16)x[i];
}

// W [K=1024][N=1024] f32  ->  Wt [N][K] f16
__global__ void transpose_to_f16(const float* __restrict__ W,
                                 _Float16* __restrict__ Wt) {
  int idx = blockIdx.x * 256 + threadIdx.x;     // 1M elements
  int k = idx >> 10, n = idx & 1023;
  Wt[(size_t)n * DD + k] = (_Float16)W[idx];
}

// -------------------------------------------------------------- WMMA GEMM ---
// C[M,N] = A[M,K] * Bt[N,K]^T ; block = 4 waves, block tile 64(M) x 64(N),
// each wave: 16x64 slice, 4 WMMAs per K=32 step.
// MODE 0: f16 out, head-split  [B,H,S,DH]    (Q, K)
// MODE 1: f16 out, head-split transposed [B,H,DH,S]  (V^T)
// MODE 2: f16 out, flat, relu            (FFN hidden)
// MODE 3: f32 out, flat, + residual      (W1/W3 outputs)
template <int MODE>
__global__ __launch_bounds__(128) void gemm16(const _Float16* __restrict__ A,
                                              const _Float16* __restrict__ Bt,
                                              void* __restrict__ outp,
                                              const float* __restrict__ resid) {
  const int lane  = threadIdx.x & 31;
  const int wv    = threadIdx.x >> 5;
  const int tileM = blockIdx.y * 64 + wv * 16;
  const int tileN = blockIdx.x * 64;
  const int mloc  = lane & 15;
  const int agrp  = (lane < 16) ? 0 : 16;
  const int bgrp  = (lane < 16) ? 0 : 16;
  const int rofs  = (lane < 16) ? 0 : 8;

  const _Float16* arow = A + (size_t)(tileM + mloc) * DD;
  const _Float16* b0 = Bt + (size_t)(tileN +  0 + mloc) * DD + bgrp;
  const _Float16* b1 = Bt + (size_t)(tileN + 16 + mloc) * DD + bgrp;
  const _Float16* b2 = Bt + (size_t)(tileN + 32 + mloc) * DD + bgrp;
  const _Float16* b3 = Bt + (size_t)(tileN + 48 + mloc) * DD + bgrp;

  v8f acc[4] = {};
  for (int k = 0; k < DD; k += 32) {
    if (k + 256 < DD) __builtin_prefetch(arow + k + 256, 0, 1);
    const v16h af = load_frag_split(arow + k + agrp);
    acc[0] = wmma_f16(af, load_frag_contig(b0 + k), acc[0]);
    acc[1] = wmma_f16(af, load_frag_contig(b1 + k), acc[1]);
    acc[2] = wmma_f16(af, load_frag_contig(b2 + k), acc[2]);
    acc[3] = wmma_f16(af, load_frag_contig(b3 + k), acc[3]);
  }

#pragma unroll
  for (int t = 0; t < 4; ++t) {
#pragma unroll
    for (int r = 0; r < 8; ++r) {
      const int grow = tileM + r + rofs;
      const int gcol = tileN + t * 16 + mloc;
      const float v = acc[t][r];
      if constexpr (MODE == 0) {
        const int b_ = grow >> 11, s_ = grow & (SS - 1);
        const int h_ = gcol >> 6,  d_ = gcol & (DHD - 1);
        ((_Float16*)outp)[(((size_t)(b_ * HH + h_)) * SS + s_) * DHD + d_] =
            (_Float16)v;
      } else if constexpr (MODE == 1) {
        const int b_ = grow >> 11, s_ = grow & (SS - 1);
        const int h_ = gcol >> 6,  d_ = gcol & (DHD - 1);
        ((_Float16*)outp)[(((size_t)(b_ * HH + h_)) * DHD + d_) * SS + s_] =
            (_Float16)v;
      } else if constexpr (MODE == 2) {
        ((_Float16*)outp)[(size_t)grow * DD + gcol] = (_Float16)fmaxf(v, 0.0f);
      } else {
        const size_t idx = (size_t)grow * DD + gcol;
        ((float*)outp)[idx] = v + resid[idx];
      }
    }
  }
}

// -------------------------------------------------------- flash attention ---
// One wave handles 16 query rows for one (b,h). Online softmax over key
// blocks of 32. Q,K in [B,H,S,DH] f16 ; V transposed [B,H,DH,S] f16.
// Softmax statistics are tracked in the log2 domain with the 1/sqrt(DH)
// scale pre-folded (SCL), so the exp path is a single v_exp per element.
__global__ __launch_bounds__(128) void attn_fwd(const _Float16* __restrict__ Q,
                                                const _Float16* __restrict__ K,
                                                const _Float16* __restrict__ V,
                                                _Float16* __restrict__ ctx) {
  __shared__ __align__(16) _Float16 pst[4][16][32];
  const int lane = threadIdx.x & 31;
  const int wv   = threadIdx.x >> 5;
  const int wid  = blockIdx.x * 4 + wv;
  const int qt = wid & (SS / 16 - 1);
  const int h  = (wid >> 7) & (HH - 1);
  const int b  = wid >> 11;
  const size_t hoff = ((size_t)(b * HH + h)) * SS * DHD;
  const _Float16* qb = Q + hoff;
  const _Float16* kb = K + hoff;
  const _Float16* vb = V + hoff;  // [DH][S] for this head, same head stride
  const int mloc = lane & 15;
  const int agrp = (lane < 16) ? 0 : 16;
  const int bgrp = (lane < 16) ? 0 : 16;
  const int rofs = (lane < 16) ? 0 : 8;

  const _Float16* qrow = qb + (size_t)(qt * 16 + mloc) * DHD;
  const v16h qf0 = load_frag_split(qrow + agrp);        // dh 0..31
  const v16h qf1 = load_frag_split(qrow + 32 + agrp);   // dh 32..63

  v8f o[4] = {};
  float rmax[8], rsum[8];   // rmax in log2 domain (scores pre-scaled by SCL)
#pragma unroll
  for (int r = 0; r < 8; ++r) { rmax[r] = -3.0e38f; rsum[r] = 0.0f; }

  for (int j = 0; j < SS; j += 32) {
    v8f sc[2] = {};
#pragma unroll
    for (int t = 0; t < 2; ++t) {  // two 16-key score tiles
      const _Float16* krow = kb + (size_t)(j + t * 16 + mloc) * DHD + bgrp;
      sc[t] = wmma_f16(qf0, load_frag_contig(krow), sc[t]);
      sc[t] = wmma_f16(qf1, load_frag_contig(krow + 32), sc[t]);
    }
#pragma unroll
    for (int r = 0; r < 8; ++r) {
      const float a = sc[0][r] * SCL;       // log2-domain scaled score
      const float c = sc[1][r] * SCL;
      const float mx = rowmax16(fmaxf(a, c));
      const float nm = fmaxf(rmax[r], mx);
      const float alpha = exp2f(rmax[r] - nm);
      rmax[r] = nm;
      const float p0 = exp2f(a - nm);
      const float p1 = exp2f(c - nm);
      const float ps = rowsum16(p0 + p1);
      rsum[r] = rsum[r] * alpha + ps;
      const int row = r + rofs;
      pst[wv][row][mloc]      = (_Float16)p0;
      pst[wv][row][16 + mloc] = (_Float16)p1;
#pragma unroll
      for (int t = 0; t < 4; ++t) o[t][r] *= alpha;
    }
#if __has_builtin(__builtin_amdgcn_s_wait_dscnt)
    __builtin_amdgcn_s_wait_dscnt(0);
#else
    asm volatile("s_wait_dscnt 0" ::: "memory");
#endif
    const v16h pf = load_frag_split(&pst[wv][mloc][agrp]);  // P as A 16x32
#pragma unroll
    for (int t = 0; t < 4; ++t) {  // 4 dh tiles from V^T (contiguous keys)
      const _Float16* vrow = vb + (size_t)(t * 16 + mloc) * SS + j + bgrp;
      o[t] = wmma_f16(pf, load_frag_contig(vrow), o[t]);
    }
  }
#pragma unroll
  for (int r = 0; r < 8; ++r) {
    const float inv = 1.0f / rsum[r];
    const int srow = qt * 16 + r + rofs;
    _Float16* dst = ctx + ((size_t)b * SS + srow) * DD + h * DHD;
#pragma unroll
    for (int t = 0; t < 4; ++t) dst[t * 16 + mloc] = (_Float16)(o[t][r] * inv);
  }
}

// --------------------------------------------------------------- layernorm --
__global__ __launch_bounds__(256) void ln_kernel(const float* __restrict__ in,
                                                 const float* __restrict__ g,
                                                 const float* __restrict__ bt,
                                                 float* __restrict__ o32,
                                                 _Float16* __restrict__ o16,
                                                 int emit16) {
  const int row = blockIdx.x, tid = threadIdx.x;
  const float4 v = ((const float4*)(in + (size_t)row * DD))[tid];
  float s  = v.x + v.y + v.z + v.w;
  float sq = v.x * v.x + v.y * v.y + v.z * v.z + v.w * v.w;
  __shared__ float sh[256], sh2[256];
  sh[tid] = s; sh2[tid] = sq;
  __syncthreads();
  for (int st = 128; st > 0; st >>= 1) {
    if (tid < st) { sh[tid] += sh[tid + st]; sh2[tid] += sh2[tid + st]; }
    __syncthreads();
  }
  const float mean = sh[0] * (1.0f / DD);
  const float var  = sh2[0] * (1.0f / DD) - mean * mean;
  const float rstd = rsqrtf(var + 1e-3f);
  const float4 gg = ((const float4*)g)[tid];
  const float4 bb = ((const float4*)bt)[tid];
  float4 r;
  r.x = (v.x - mean) * rstd * gg.x + bb.x;
  r.y = (v.y - mean) * rstd * gg.y + bb.y;
  r.z = (v.z - mean) * rstd * gg.z + bb.z;
  r.w = (v.w - mean) * rstd * gg.w + bb.w;
  ((float4*)(o32 + (size_t)row * DD))[tid] = r;
  if (emit16) {
    _Float16* p = o16 + (size_t)row * DD + tid * 4;
    p[0] = (_Float16)r.x; p[1] = (_Float16)r.y;
    p[2] = (_Float16)r.z; p[3] = (_Float16)r.w;
  }
}

// ------------------------------------------------------------------ driver --
extern "C" void kernel_launch(void* const* d_in, const int* in_sizes, int n_in,
                              void* d_out, int out_size, void* d_ws,
                              size_t ws_size, hipStream_t stream) {
  (void)in_sizes; (void)n_in; (void)out_size; (void)ws_size;
  const float* q     = (const float*)d_in[0];
  const float* Wmats[6] = {(const float*)d_in[1], (const float*)d_in[2],
                           (const float*)d_in[3], (const float*)d_in[4],
                           (const float*)d_in[5], (const float*)d_in[6]};
  const float* gamma = (const float*)d_in[7];
  const float* beta  = (const float*)d_in[8];
  float* out = (float*)d_out;

  char* ws = (char*)d_ws;
  const size_t MB = 1024ull * 1024ull;
  _Float16* xf16  = (_Float16*)(ws + 0);        //  8MB (reused as FFN hidden)
  _Float16* wt    = (_Float16*)(ws + 8 * MB);   // 12MB, 6 x (1024*1024) f16
  _Float16* qh    = (_Float16*)(ws + 20 * MB);  //  8MB [B,H,S,DH]
  _Float16* kh    = (_Float16*)(ws + 28 * MB);  //  8MB [B,H,S,DH]
  _Float16* vt    = (_Float16*)(ws + 36 * MB);  //  8MB [B,H,DH,S]
  _Float16* ctx   = (_Float16*)(ws + 44 * MB);  //  8MB [B*S, D]
  float*    tmp32 = (float*)(ws + 52 * MB);     // 16MB (W1 out, then W3 out)
  float*    res32 = (float*)(ws + 68 * MB);     // 16MB
  _Float16* res16 = (_Float16*)(ws + 84 * MB);  //  8MB
  _Float16* h16   = xf16;                       // reuse after QKV GEMMs

  _Float16* wtq = wt + 0ull * DD * DD;
  _Float16* wtk = wt + 1ull * DD * DD;
  _Float16* wtv = wt + 2ull * DD * DD;
  _Float16* wt1 = wt + 3ull * DD * DD;
  _Float16* wt2 = wt + 4ull * DD * DD;
  _Float16* wt3 = wt + 5ull * DD * DD;

  // 1) convert activations + weights (weights transposed to [N][K])
  const int nact = MTOT * DD;
  cvt_f32_to_f16<<<(nact + 255) / 256, 256, 0, stream>>>(q, xf16, nact);
  _Float16* wdst[6] = {wtq, wtk, wtv, wt1, wt2, wt3};
  for (int i = 0; i < 6; ++i)
    transpose_to_f16<<<(DD * DD + 255) / 256, 256, 0, stream>>>(Wmats[i],
                                                                wdst[i]);

  // 2) Q/K/V projections
  dim3 ggrid(DD / 64, MTOT / 64);
  gemm16<0><<<ggrid, 128, 0, stream>>>(xf16, wtq, qh, nullptr);
  gemm16<0><<<ggrid, 128, 0, stream>>>(xf16, wtk, kh, nullptr);
  gemm16<1><<<ggrid, 128, 0, stream>>>(xf16, wtv, vt, nullptr);

  // 3) attention: B*H*(S/16) = 4096 waves, 4 waves/block
  attn_fwd<<<1024, 128, 0, stream>>>(qh, kh, vt, ctx);

  // 4) W1 + residual(query) -> LN1 (emits f32 + f16 residual)
  gemm16<3><<<ggrid, 128, 0, stream>>>(ctx, wt1, tmp32, q);
  ln_kernel<<<MTOT, 256, 0, stream>>>(tmp32, gamma, beta, res32, res16, 1);

  // 5) FFN: relu(res @ W2) @ W3 + res -> LN2 -> out
  gemm16<2><<<ggrid, 128, 0, stream>>>(res16, wt2, h16, nullptr);
  gemm16<3><<<ggrid, 128, 0, stream>>>(h16, wt3, tmp32, res32);
  ln_kernel<<<MTOT, 256, 0, stream>>>(tmp32, gamma, beta, out, nullptr, 0);
}
